// EncoderLayer_64587718197946
// MI455X (gfx1250) — matrix operand order repaired and checked by reading, not compiled, not run on hardware
//
#include <hip/hip_runtime.h>

// ---------- types ----------
typedef __attribute__((ext_vector_type(16))) __bf16 v16bf;
typedef __attribute__((ext_vector_type(8)))  float  v8f;
typedef __attribute__((ext_vector_type(4)))  float  f32x4;
typedef __attribute__((ext_vector_type(4)))  unsigned int u32x4;
typedef __attribute__((ext_vector_type(2)))  unsigned int u32x2;
typedef __attribute__((ext_vector_type(8)))  int i32x8;
typedef __attribute__((ext_vector_type(4)))  int i32x4;

union FragBF {            // 16 bf16 = 32 bytes = 8 VGPRs (one WMMA A/B operand)
    v16bf v;
    u32x4 u[2];
};

__device__ __forceinline__ unsigned short f2bf(float f) {
    unsigned int u = __float_as_uint(f);
    unsigned int r = u + 0x7FFFu + ((u >> 16) & 1u);   // round-to-nearest-even
    return (unsigned short)(r >> 16);
}

__device__ __forceinline__ v8f wmma_bf16(const v16bf& a, const v16bf& b, v8f c) {
    return __builtin_amdgcn_wmma_f32_16x16x32_bf16(false, a, false, b, (short)0, c, false, false);
}

#if defined(__has_builtin)
#if __has_builtin(__builtin_amdgcn_tensor_load_to_lds) && __has_builtin(__builtin_amdgcn_s_wait_tensorcnt)
#define HAVE_TDM 1
#endif
#endif

#ifdef HAVE_TDM
// ---------- Tensor Data Mover: async 2D tile (bf16) -> LDS with HW padding ----------
// D# layout per CDNA5 ISA (2D: groups 2/3 NULL-equivalent zeros).
__device__ __forceinline__ void tdm_load_tile_2d(
    unsigned lds_byte_addr,        // LDS byte offset of tile start
    const void* gaddr,             // global address of tile start
    unsigned tensor_d0,            // tensor extent, dim0 (elements)
    unsigned tensor_d1,            // tensor extent, dim1 (elements)
    unsigned tile_d0,              // tile extent, dim0 (contiguous, elements)
    unsigned tile_d1,              // tile extent, dim1 (elements)
    unsigned long d0_stride,       // dim0 stride (elements)
    unsigned pad_interval,         // enum: 0=2DW,1=4,2=8,3=16,4=32,5=64,...
    unsigned pad_amount)           // enum: 0=1DW,1=2,...,3=4DW,...
{
    const unsigned long ga = (unsigned long)gaddr;
    u32x4 g0;
    g0[0] = 1u;                                            // count=1, user descriptor
    g0[1] = lds_byte_addr;                                 // lds_addr
    g0[2] = (unsigned)(ga & 0xFFFFFFFFu);                  // global_addr[31:0]
    g0[3] = (unsigned)((ga >> 32) & 0x01FFFFFFu)           // global_addr[56:32]
          | (2u << 30);                                    // type = 2 ("image")
    i32x8 g1;
    g1[0] = (int)((1u << 16)                               // data_size = 1 -> 2 bytes
                | (1u << 20)                               // pad_enable
                | (pad_interval << 22)
                | (pad_amount << 25));
    g1[1] = (int)((tensor_d0 & 0xFFFFu) << 16);            // tensor_dim0[15:0]
    g1[2] = (int)((tensor_d0 >> 16) | ((tensor_d1 & 0xFFFFu) << 16));
    g1[3] = (int)((tensor_d1 >> 16) | (tile_d0 << 16));    // tensor_dim1[31:16], tile_dim0
    g1[4] = (int)(tile_d1 & 0xFFFFu);                      // tile_dim1 (tile_dim2 = 0)
    g1[5] = (int)(unsigned)(d0_stride & 0xFFFFFFFFu);      // tensor_dim0_stride[31:0]
    g1[6] = (int)(unsigned)((d0_stride >> 32) & 0xFFFFu);  // tensor_dim0_stride[47:32]
    g1[7] = 0;                                             // tensor_dim1_stride (unused, 2D)
    const i32x4 z4 = {0, 0, 0, 0};
#if __clang_major__ >= 23
    const i32x8 z8 = {0, 0, 0, 0, 0, 0, 0, 0};
    __builtin_amdgcn_tensor_load_to_lds(g0, g1, z4, z4, z8, 0);
#else
    __builtin_amdgcn_tensor_load_to_lds(g0, g1, z4, z4, 0);
#endif
}
#endif

// ---------- fp32 -> bf16 cast ----------
__global__ __launch_bounds__(256) void cast_f32_bf16(const float* __restrict__ in,
                                                     unsigned short* __restrict__ out,
                                                     int n) {
    int i  = blockIdx.x * 256 + threadIdx.x;
    int i4 = i * 4;
    if (i4 + 3 < n) {
        f32x4 f = *(const f32x4*)(in + i4);
        u32x2 p;
        p[0] = (unsigned)f2bf(f[0]) | ((unsigned)f2bf(f[1]) << 16);
        p[1] = (unsigned)f2bf(f[2]) | ((unsigned)f2bf(f[3]) << 16);
        *(u32x2*)(out + i4) = p;
    }
}

// ---------- bf16 GEMM: C[M,N] = act(A[M,K] @ B[K,N] + bias) ----------
// 256 threads = 8 waves; block tile 128x128; K step 32; TDM double-buffered tiles.
// Wave w: rows 32*(w>>1), cols 64*(w&1); 2x4 grid of 16x16 WMMA subtiles.
template <int WF, int WB, int RELU>
__global__ __launch_bounds__(256) void gemm_bf16(
    const unsigned short* __restrict__ A, int lda,
    const unsigned short* __restrict__ B, int ldb,
    const float* __restrict__ bias,
    float* __restrict__ outF, unsigned short* __restrict__ outB, int ldc,
    int K)
{
    constexpr int AST = 40;    // 128x32 A tile rows padded 64B->80B (HW pad 4 DW / 16 DW)
    constexpr int BST = 136;   // 32x128 B tile rows padded 256B->272B (HW pad 4 DW / 64 DW)
    __shared__ unsigned short As[2][128 * AST];
    __shared__ unsigned short Bs[2][32 * BST];

    const int tid  = threadIdx.x;
    const int lane = tid & 31, w = tid >> 5;
    const int lm   = lane & 15, hf = lane >> 4;
    const int wr   = w >> 1,    wc = w & 1;
    const long m0  = (long)blockIdx.x * 128;
    const long n0  = (long)blockIdx.y * 128;

    v8f c[2][4];
#pragma unroll
    for (int i = 0; i < 2; ++i)
#pragma unroll
        for (int j = 0; j < 4; ++j) c[i][j] = {};

    const int kb = hf * 8;     // A-frag K chunk base per lane-half

#ifdef HAVE_TDM
    // ---- TDM path: wave 0 drives the Tensor Data Mover, double buffered ----
    if (w == 0) {
        tdm_load_tile_2d((unsigned)(size_t)&As[0][0], A + m0 * lda, (unsigned)lda,
                         8192u, 32u, 128u, (unsigned long)lda, 3u, 3u);
        tdm_load_tile_2d((unsigned)(size_t)&Bs[0][0], B + n0, (unsigned)ldb,
                         (unsigned)K, 128u, 32u, (unsigned long)ldb, 5u, 3u);
    }
    int buf = 0;
    for (int k0 = 0; k0 < K; k0 += 32) {
        if (w == 0) {
            if (k0 + 32 < K) {
                tdm_load_tile_2d((unsigned)(size_t)&As[buf ^ 1][0],
                                 A + m0 * lda + (k0 + 32), (unsigned)lda,
                                 8192u, 32u, 128u, (unsigned long)lda, 3u, 3u);
                tdm_load_tile_2d((unsigned)(size_t)&Bs[buf ^ 1][0],
                                 B + (long)(k0 + 32) * ldb + n0, (unsigned)ldb,
                                 (unsigned)K, 128u, 32u, (unsigned long)ldb, 5u, 3u);
                __builtin_amdgcn_s_wait_tensorcnt(2);   // current tile's 2 ops done
            } else {
                __builtin_amdgcn_s_wait_tensorcnt(0);
            }
        }
        __syncthreads();   // publish current buffer to all waves
#else
    // ---- fallback path: cooperative staging ----
    const int ar = tid >> 1, ah = (tid & 1) * 16;
    const int bkr = tid >> 3, bseg = (tid & 7) * 16;
    int buf = 0;
    for (int k0 = 0; k0 < K; k0 += 32) {
        __syncthreads();
        {
            const u32x4* ag = (const u32x4*)(A + (m0 + ar) * lda + k0 + ah);
            *(u32x4*)(&As[0][ar * AST + ah])     = ag[0];
            *(u32x4*)(&As[0][ar * AST + ah + 8]) = ag[1];
            const u32x4* bg = (const u32x4*)(B + (long)(k0 + bkr) * ldb + n0 + bseg);
            *(u32x4*)(&Bs[0][bkr * BST + bseg])     = bg[0];
            *(u32x4*)(&Bs[0][bkr * BST + bseg + 8]) = bg[1];
        }
        if (k0 + 32 < K) {
            __builtin_prefetch(A + (m0 + ar) * lda + (k0 + 32) + ah, 0, 0);
            __builtin_prefetch(B + (long)(k0 + 32 + bkr) * ldb + n0 + bseg, 0, 0);
        }
        __syncthreads();
#endif

        FragBF a[2], b[4];
#pragma unroll
        for (int i = 0; i < 2; ++i) {
            const int row = wr * 32 + i * 16 + lm;     // A: per-lane row, K in chunks
            a[i].u[0] = *(const u32x4*)(&As[buf][row * AST + kb]);
            a[i].u[1] = *(const u32x4*)(&As[buf][row * AST + kb + 16]);
        }
#pragma unroll
        for (int j = 0; j < 4; ++j) {
            const int nc = wc * 64 + j * 16 + lm;      // B: N col = lane%16, K in chunks (same as A)
#pragma unroll
            for (int s = 0; s < 16; ++s)
                ((unsigned short*)&b[j])[s] = Bs[buf][(kb + (s & 7) + ((s >> 3) << 4)) * BST + nc];
        }
#pragma unroll
        for (int i = 0; i < 2; ++i)
#pragma unroll
            for (int j = 0; j < 4; ++j)
                c[i][j] = wmma_bf16(a[i].v, b[j].v, c[i][j]);

#ifdef HAVE_TDM
        __syncthreads();   // all waves done reading buffer before TDM may overwrite it
        buf ^= 1;
    }
#else
    }
#endif

    // epilogue: D layout -> row = r + 8*hf, col = lane%16
#pragma unroll
    for (int i = 0; i < 2; ++i)
#pragma unroll
        for (int j = 0; j < 4; ++j) {
            const long col = n0 + wc * 64 + j * 16 + lm;
            const float bv = bias[col];
#pragma unroll
            for (int r = 0; r < 8; ++r) {
                const long row = m0 + wr * 32 + i * 16 + r + 8 * hf;
                float vv = c[i][j][r] + bv;
                if (RELU) vv = fmaxf(vv, 0.0f);
                if (WF) outF[row * ldc + col] = vv;
                if (WB) outB[row * ldc + col] = f2bf(vv);
            }
        }
}

// ---------- flash attention, bf16 WMMA, fp32 online softmax ----------
// grid = (S/128, NH, B); 256 threads = 8 waves; wave = 16 query rows.
__global__ __launch_bounds__(256) void attention_fa(
    const unsigned short* __restrict__ qb,
    const unsigned short* __restrict__ kbuf,
    const unsigned short* __restrict__ vbuf,
    unsigned short* __restrict__ attn,
    int S)
{
    constexpr int E = 1024;
    constexpr int KTST = 40, VST = 72, PST = 40;
    __shared__ unsigned short Kt[64 * KTST];       // K^T block: [dim 64][key 32]
    __shared__ unsigned short Vs[32 * VST];        // V  block: [key 32][dim 64]
    __shared__ unsigned short Ps[8 * 16 * PST];    // per-wave P repack scratch

    const int tid = threadIdx.x, lane = tid & 31, w = tid >> 5;
    const int lm = lane & 15, hf = lane >> 4;
    const int h = blockIdx.y, bidx = blockIdx.z;
    const long rowbase = (long)bidx * S;
    const int q0 = blockIdx.x * 128 + w * 16;
    const int hd0 = h * 64;
    const int kbb = hf * 8;

    // Q strip (16 x 64) as two A fragments held in registers
    FragBF qa[2];
    {
        const unsigned short* qp = qb + (rowbase + q0 + lm) * E + hd0;
#pragma unroll
        for (int cc = 0; cc < 2; ++cc) {
            qa[cc].u[0] = *(const u32x4*)(qp + cc * 32 + kbb);
            qa[cc].u[1] = *(const u32x4*)(qp + cc * 32 + kbb + 16);
        }
    }

    v8f O[4];
#pragma unroll
    for (int jo = 0; jo < 4; ++jo) O[jo] = {};
    float m[8], l[8];
#pragma unroll
    for (int r = 0; r < 8; ++r) { m[r] = -3.0e38f; l[r] = 0.0f; }

    const int key = tid >> 3, d0 = (tid & 7) * 8;  // cooperative K/V staging
    unsigned short* psw = Ps + w * 16 * PST;

    for (int j0 = 0; j0 < S; j0 += 32) {
        __syncthreads();
        {   // stage K (transposed) and V block
            const long krow = rowbase + j0 + key;
            u32x4 kv = *(const u32x4*)(kbuf + krow * E + hd0 + d0);
            const unsigned short* ks = (const unsigned short*)&kv;
#pragma unroll
            for (int i = 0; i < 8; ++i) Kt[(d0 + i) * KTST + key] = ks[i];
            u32x4 vv = *(const u32x4*)(vbuf + krow * E + hd0 + d0);
            *(u32x4*)(&Vs[key * VST + d0]) = vv;
        }
        __syncthreads();

        // scores S = Q K^T : 2 key subtiles x (2 WMMA over head-dim 64)
        v8f s[2];
#pragma unroll
        for (int jn = 0; jn < 2; ++jn) {
            s[jn] = {};
#pragma unroll
            for (int cc = 0; cc < 2; ++cc) {
                FragBF kf;   // B frag: N = key lane%16, K (dim) in chunks (same as A)
#pragma unroll
                for (int s = 0; s < 16; ++s)
                    ((unsigned short*)&kf)[s] = Kt[(cc * 32 + kbb + (s & 7) + ((s >> 3) << 4)) * KTST + jn * 16 + lm];
                s[jn] = wmma_bf16(qa[cc].v, kf.v, s[jn]);
            }
        }

        // online softmax per query row (rows live in VGPR index + lane-half)
#pragma unroll
        for (int r = 0; r < 8; ++r) {
            float t0 = s[0][r] * 0.125f;   // 1/sqrt(64)
            float t1 = s[1][r] * 0.125f;
            float mx = fmaxf(t0, t1);
#pragma unroll
            for (int o = 1; o < 16; o <<= 1) mx = fmaxf(mx, __shfl_xor(mx, o, 32));
            const float mn = fmaxf(m[r], mx);
            const float al = __expf(m[r] - mn);
            const float p0 = __expf(t0 - mn);
            const float p1 = __expf(t1 - mn);
            float sm = p0 + p1;
#pragma unroll
            for (int o = 1; o < 16; o <<= 1) sm += __shfl_xor(sm, o, 32);
            l[r] = l[r] * al + sm;
            m[r] = mn;
#pragma unroll
            for (int jo = 0; jo < 4; ++jo) O[jo][r] *= al;
            psw[(r + 8 * hf) * PST + lm]      = f2bf(p0);
            psw[(r + 8 * hf) * PST + 16 + lm] = f2bf(p1);
        }

        // same-wave LDS ops are in-order; make the RAW explicit
        asm volatile("s_wait_dscnt 0" ::: "memory");

        FragBF pa;   // P repacked into A-fragment layout (16 rows x 32 keys)
        pa.u[0] = *(const u32x4*)(&psw[lm * PST + kbb]);
        pa.u[1] = *(const u32x4*)(&psw[lm * PST + kbb + 16]);
#pragma unroll
        for (int jo = 0; jo < 4; ++jo) {
            FragBF vf;   // B frag: N = head dim lane%16, K (key) in chunks (same as A)
#pragma unroll
            for (int s = 0; s < 16; ++s)
                ((unsigned short*)&vf)[s] = Vs[(kbb + (s & 7) + ((s >> 3) << 4)) * VST + jo * 16 + lm];
            O[jo] = wmma_bf16(pa.v, vf.v, O[jo]);
        }
    }

    // normalize and store bf16 (concat-head layout for the Wo GEMM)
#pragma unroll
    for (int r = 0; r < 8; ++r) {
        const float inv = 1.0f / l[r];
        const long row = rowbase + q0 + r + 8 * hf;
#pragma unroll
        for (int jo = 0; jo < 4; ++jo)
            attn[row * E + hd0 + jo * 16 + lm] = f2bf(O[jo][r] * inv);
    }
}

// ---------- residual + LayerNorm (ddof=1, divide by std+eps) ----------
__global__ __launch_bounds__(256) void add_layernorm(
    const float* __restrict__ xin, const float* __restrict__ rin,
    const float* __restrict__ g, const float* __restrict__ be,
    float* __restrict__ outF, unsigned short* __restrict__ outB)
{
    __shared__ float red[8];
    const int tid = threadIdx.x, lane = tid & 31, w = tid >> 5;
    const long row = blockIdx.x;
    const int c0 = tid * 4;

    f32x4 a  = *(const f32x4*)(xin + row * 1024 + c0);
    f32x4 rr = *(const f32x4*)(rin + row * 1024 + c0);
    float v[4];
#pragma unroll
    for (int i = 0; i < 4; ++i) v[i] = a[i] + rr[i];

    float s = v[0] + v[1] + v[2] + v[3];
#pragma unroll
    for (int o = 16; o > 0; o >>= 1) s += __shfl_xor(s, o, 32);
    if (lane == 0) red[w] = s;
    __syncthreads();
    float tot = 0;
#pragma unroll
    for (int i = 0; i < 8; ++i) tot += red[i];
    const float mean = tot * (1.0f / 1024.0f);

    float qs = 0;
#pragma unroll
    for (int i = 0; i < 4; ++i) { const float d = v[i] - mean; qs += d * d; }
    __syncthreads();
#pragma unroll
    for (int o = 16; o > 0; o >>= 1) qs += __shfl_xor(qs, o, 32);
    if (lane == 0) red[w] = qs;
    __syncthreads();
    float tot2 = 0;
#pragma unroll
    for (int i = 0; i < 8; ++i) tot2 += red[i];
    const float rs = 1.0f / (sqrtf(tot2 * (1.0f / 1023.0f)) + 1e-6f);

    f32x4 gg = *(const f32x4*)(g + c0);
    f32x4 bb = *(const f32x4*)(be + c0);
    f32x4 y;
#pragma unroll
    for (int i = 0; i < 4; ++i) y[i] = gg[i] * ((v[i] - mean) * rs) + bb[i];
    *(f32x4*)(outF + row * 1024 + c0) = y;
    if (outB) {
        u32x2 p;
        p[0] = (unsigned)f2bf(y[0]) | ((unsigned)f2bf(y[1]) << 16);
        p[1] = (unsigned)f2bf(y[2]) | ((unsigned)f2bf(y[3]) << 16);
        *(u32x2*)(outB + row * 1024 + c0) = p;
    }
}

// ---------- launcher ----------
extern "C" void kernel_launch(void* const* d_in, const int* in_sizes, int n_in,
                              void* d_out, int out_size, void* d_ws, size_t ws_size,
                              hipStream_t stream) {
    const float* x   = (const float*)d_in[0];
    const float* Wq  = (const float*)d_in[1];  const float* bq  = (const float*)d_in[2];
    const float* Wk  = (const float*)d_in[3];  const float* bk  = (const float*)d_in[4];
    const float* Wv  = (const float*)d_in[5];  const float* bv  = (const float*)d_in[6];
    const float* Wo  = (const float*)d_in[7];  const float* bo  = (const float*)d_in[8];
    const float* W1  = (const float*)d_in[9];  const float* b1  = (const float*)d_in[10];
    const float* W2  = (const float*)d_in[11]; const float* b2  = (const float*)d_in[12];
    const float* g1  = (const float*)d_in[13]; const float* be1 = (const float*)d_in[14];
    const float* g2  = (const float*)d_in[15]; const float* be2 = (const float*)d_in[16];

    constexpr long M = 8192;         // B*S
    constexpr int  E = 1024, HID = 4096, S = 2048, NH = 16, Bb = 4;

    char* ws = (char*)d_ws;
    size_t off = 0;
    auto alloc = [&](size_t bytes) -> char* {
        char* p = ws + off;
        off = (off + bytes + 255) & ~(size_t)255;
        return p;
    };

    unsigned short* xb   = (unsigned short*)alloc(M * E * 2);
    unsigned short* wqb  = (unsigned short*)alloc((size_t)E * E * 2);
    unsigned short* wkb  = (unsigned short*)alloc((size_t)E * E * 2);
    unsigned short* wvb  = (unsigned short*)alloc((size_t)E * E * 2);
    unsigned short* wob  = (unsigned short*)alloc((size_t)E * E * 2);
    unsigned short* w1b  = (unsigned short*)alloc((size_t)E * HID * 2);
    unsigned short* w2b  = (unsigned short*)alloc((size_t)HID * E * 2);
    unsigned short* qbuf = (unsigned short*)alloc(M * E * 2);
    unsigned short* kbf  = (unsigned short*)alloc(M * E * 2);
    unsigned short* vbf  = (unsigned short*)alloc(M * E * 2);
    unsigned short* atb  = (unsigned short*)alloc(M * E * 2);
    float*          proj = (float*)alloc(M * E * 4);
    float*          h1f  = (float*)alloc(M * E * 4);
    unsigned short* h1b  = (unsigned short*)alloc(M * E * 2);
    unsigned short* ffh  = (unsigned short*)alloc(M * HID * 2);
    float*          ff2  = (float*)alloc(M * E * 4);

    auto cast = [&](const float* src, unsigned short* dst, long n) {
        cast_f32_bf16<<<dim3((unsigned)(n / 1024)), dim3(256), 0, stream>>>(src, dst, (int)n);
    };
    cast(x,  xb,  M * E);
    cast(Wq, wqb, (long)E * E);
    cast(Wk, wkb, (long)E * E);
    cast(Wv, wvb, (long)E * E);
    cast(Wo, wob, (long)E * E);
    cast(W1, w1b, (long)E * HID);
    cast(W2, w2b, (long)HID * E);

    // QKV projections (bf16 out)
    gemm_bf16<0, 1, 0><<<dim3(64, 8), 256, 0, stream>>>(xb, E, wqb, E, bq, nullptr, qbuf, E, E);
    gemm_bf16<0, 1, 0><<<dim3(64, 8), 256, 0, stream>>>(xb, E, wkb, E, bk, nullptr, kbf,  E, E);
    gemm_bf16<0, 1, 0><<<dim3(64, 8), 256, 0, stream>>>(xb, E, wvb, E, bv, nullptr, vbf,  E, E);

    // flash attention
    attention_fa<<<dim3(S / 128, NH, Bb), 256, 0, stream>>>(qbuf, kbf, vbf, atb, S);

    // output projection (fp32 out)
    gemm_bf16<1, 0, 0><<<dim3(64, 8), 256, 0, stream>>>(atb, E, wob, E, bo, proj, nullptr, E, E);

    // residual + LN1 -> fp32 + bf16
    add_layernorm<<<dim3((unsigned)M), 256, 0, stream>>>(x, proj, g1, be1, h1f, h1b);

    // FFN
    gemm_bf16<0, 1, 1><<<dim3(64, 32), 256, 0, stream>>>(h1b, E, w1b, HID, b1, nullptr, ffh, HID, E);
    gemm_bf16<1, 0, 0><<<dim3(64, 8),  256, 0, stream>>>(ffh, HID, w2b, E, b2, ff2, nullptr, E, HID);

    // residual + LN2 -> final fp32 output
    add_layernorm<<<dim3((unsigned)M), 256, 0, stream>>>(h1f, ff2, g2, be2, (float*)d_out, nullptr);
}
